// VideoPrismMultiheadAttentionPoolingHead_7395933683759
// MI455X (gfx1250) — matrix operand order repaired and checked by reading, not compiled, not run on hardware
//
#include <hip/hip_runtime.h>

#define HIDDEN   768
#define INTER    3072
#define HEADS    12
#define HEAD_DIM 256
#define BATCH    8
#define SEQ      4096

typedef __attribute__((ext_vector_type(16))) __bf16        v16bf;
typedef __attribute__((ext_vector_type(8)))  __bf16        v8bf;
typedef __attribute__((ext_vector_type(8)))  float         v8f;
typedef __attribute__((ext_vector_type(8)))  unsigned short v8us;

// ---- workspace layout (bytes) ----
#define WS_QS      ((size_t)0)         // 3072 f32 : scaled query
#define WS_WKE     ((size_t)16384)     // 768*16 f32 : folded Wk (heads padded to 16)
#define WS_PACK    ((size_t)65536)     // 24*32*16 bf16 : WMMA B fragments of wk_eff
#define WS_CK      ((size_t)98304)     // 16 f32 : per-head logit bias (bk . qs)
#define WS_LOGITS  ((size_t)131072)    // B*S*16 f32 : logits, [b][s][h16]
#define WS_ATTNBF  ((size_t)4194304)   // B*16*S bf16 : attn as WMMA A rows, [b][h16][s]
#define WS_U       ((size_t)5242880)   // B*16*768 f32 : u = attn^T @ HS
#define WS_CTX     ((size_t)6291456)   // B*3072 f32
#define WS_HB      ((size_t)8388608)   // B*S*768 bf16 : hidden in bf16

__device__ __forceinline__ unsigned short f32_to_bf16_bits(float f) {
  union { float f; unsigned u; } x; x.f = f;
  unsigned r = x.u + 0x7FFFu + ((x.u >> 16) & 1u);   // round-to-nearest-even
  return (unsigned short)(r >> 16);
}

__device__ __forceinline__ v16bf cat_v8bf(v8bf lo, v8bf hi) {
  v16bf r;
#pragma unroll
  for (int j = 0; j < 8; ++j) { r[j] = lo[j]; r[j + 8] = hi[j]; }
  return r;
}

__device__ __forceinline__ float softplusf(float x) {
  return (x > 20.f) ? x : log1pf(__expf(x));
}

// ---- 1) hidden f32 -> bf16 (packed 2-at-a-time) ----
__global__ void __launch_bounds__(256) k_cvt_hidden(const float* __restrict__ hs,
                                                    unsigned int* __restrict__ hb2, int n2) {
  int i = blockIdx.x * 256 + threadIdx.x;
  if (i >= n2) return;
  float2 f = ((const float2*)hs)[i];
  unsigned lo = f32_to_bf16_bits(f.x);
  unsigned hi = f32_to_bf16_bits(f.y);
  hb2[i] = lo | (hi << 16);
}

// ---- 2) qs = (pq @ Wq + bq) * scale * softplus(pds) ----
__global__ void __launch_bounds__(256) k_qproj(const float* __restrict__ pq, const float* __restrict__ Wq,
                                               const float* __restrict__ bq, const float* __restrict__ pds,
                                               float* __restrict__ qs) {
  int i = blockIdx.x * 256 + threadIdx.x;       // 0..3071
  float acc = bq[i];
  for (int c = 0; c < HIDDEN; ++c) acc += pq[c] * Wq[(size_t)c * INTER + i];
  const float SCALE = 1.442695041f / 16.0f;     // r_softplus_0 / sqrt(head_dim)
  qs[i] = acc * SCALE * softplusf(pds[i & 255]);
}

// ---- 3) wk_eff[c][h16] = sum_d Wk[c][256h+d] * qs[256h+d] ----
__global__ void __launch_bounds__(256) k_wkeff(const float* __restrict__ Wk, const float* __restrict__ qs,
                                               float* __restrict__ wke) {
  int t = blockIdx.x * 256 + threadIdx.x;       // 768*16
  int c = t >> 4, h = t & 15;
  float acc = 0.f;
  if (h < HEADS) {
    const float* wr = Wk + (size_t)c * INTER + h * HEAD_DIM;
    const float* q  = qs + h * HEAD_DIM;
    for (int d = 0; d < HEAD_DIM; ++d) acc += wr[d] * q[d];
  }
  wke[t] = acc;
}

// ---- 4) pack wk_eff into per-lane WMMA B fragments; ck[h] = bk_h . qs_h ----
__global__ void __launch_bounds__(256) k_pack(const float* __restrict__ wke, const float* __restrict__ bk,
                                              const float* __restrict__ qs, unsigned short* __restrict__ wkp,
                                              float* __restrict__ ck) {
  int t = blockIdx.x * 256 + threadIdx.x;       // 24*32*16 = 12288
  int j    = t & 15;
  int lane = (t >> 4) & 31;
  int kc   = t >> 9;
  int col  = lane & 15;
  int k    = kc * 32 + (j & 7) + ((lane >> 4) << 3) + ((j >> 3) << 4);  // B: table order, K 8h+s / 16+8h+s
  wkp[t] = f32_to_bf16_bits(wke[k * 16 + col]);
  if (t < 16) {
    float a = 0.f;
    if (t < HEADS) {
      const float* q  = qs + t * HEAD_DIM;
      const float* bb = bk + t * HEAD_DIM;
      for (int d = 0; d < HEAD_DIM; ++d) a += bb[d] * q[d];
    }
    ck[t] = a;
  }
}

// ---- 5) logits = HS(bf16) @ wk_eff + ck, via v_wmma_f32_16x16x32_bf16 ----
__global__ void __launch_bounds__(256) k_logits(const unsigned short* __restrict__ hb,
                                                const unsigned short* __restrict__ wkp,
                                                const float* __restrict__ ck,
                                                float* __restrict__ logits) {
  int wave = threadIdx.x >> 5;
  int lane = threadIdx.x & 31;
  int tile = blockIdx.x * 8 + wave;             // 0..2047 (B*S/16)
  int b  = tile >> 8;
  int s0 = (tile & 255) << 4;
  int row  = lane & 15;
  int koff = (lane >> 4) << 3;                  // A: 0 or 8
  v8f acc = {0.f, 0.f, 0.f, 0.f, 0.f, 0.f, 0.f, 0.f};
  const unsigned short* arow = hb + ((size_t)(b * SEQ + s0 + row)) * HIDDEN;
#pragma unroll 4
  for (int kc = 0; kc < 24; ++kc) {
    int cb = kc * 32 + koff;
    v8bf alo = *(const v8bf*)(arow + cb);
    v8bf ahi = *(const v8bf*)(arow + cb + 16);
    v16bf a  = cat_v8bf(alo, ahi);
    v16bf bf = *(const v16bf*)(wkp + ((size_t)(kc * 32 + lane)) * 16);
    acc = __builtin_amdgcn_wmma_f32_16x16x32_bf16(false, a, false, bf, (short)0, acc, false, false);
  }
  int col = lane & 15;
  int rb  = (lane >> 4) << 3;                   // D: rows r / r+8
  float ckv = ck[col];
#pragma unroll
  for (int r = 0; r < 8; ++r)
    logits[((size_t)(b * SEQ + s0 + rb + r)) * 16 + col] = acc[r] + ckv;
}

// ---- 6) softmax over s per (b,h); emit attn (f32 out) + bf16 A-fragment rows ----
__global__ void __launch_bounds__(256) k_softmax(const float* __restrict__ logits,
                                                 float* __restrict__ attn_out,
                                                 unsigned short* __restrict__ attnbf) {
  int b = blockIdx.x >> 4;
  int h = blockIdx.x & 15;
  int t = threadIdx.x;
  unsigned short* abrow = attnbf + ((size_t)(b * 16 + h)) * SEQ;
  if (h >= HEADS) {                              // padded head rows contribute zero
    for (int s = t; s < SEQ; s += 256) abrow[s] = 0;
    return;
  }
  __shared__ float red[256];
  float vals[16];
  float mx = -3.4e38f;
#pragma unroll
  for (int k = 0; k < 16; ++k) {
    int s = t * 16 + k;
    vals[k] = logits[((size_t)(b * SEQ + s)) * 16 + h];
    mx = fmaxf(mx, vals[k]);
  }
  red[t] = mx; __syncthreads();
  for (int o = 128; o > 0; o >>= 1) { if (t < o) red[t] = fmaxf(red[t], red[t + o]); __syncthreads(); }
  float m = red[0]; __syncthreads();
  float sum = 0.f;
#pragma unroll
  for (int k = 0; k < 16; ++k) { vals[k] = __expf(vals[k] - m); sum += vals[k]; }
  red[t] = sum; __syncthreads();
  for (int o = 128; o > 0; o >>= 1) { if (t < o) red[t] += red[t + o]; __syncthreads(); }
  float inv = 1.f / red[0];
  float* orow = attn_out + ((size_t)(b * HEADS + h)) * SEQ;
#pragma unroll
  for (int k = 0; k < 16; ++k) {
    int s = t * 16 + k;
    float a = vals[k] * inv;
    orow[s]  = a;
    abrow[s] = f32_to_bf16_bits(a);
  }
}

// ---- 7) u[b] = attn(16xS) @ HS(SxH) via WMMA; B staged through LDS transpose ----
// One wave per block. Double-buffered 32(s) x 16(c) bf16 tile in LDS:
//   global: coalesced 16B loads (2 per lane)   ->  tileT[c_local][s_local]
//   fragment: lane L reads 8+8 bf16 at s 8h, 16+8h  ->  two ds_load_b128
__global__ void __launch_bounds__(32) k_ugemm(const unsigned short* __restrict__ hb,
                                              const unsigned short* __restrict__ attnbf,
                                              float* __restrict__ u) {
  __shared__ __align__(16) unsigned short tileT[2][16][32];   // [buf][c_local][s_local]
  int lane = threadIdx.x;
  int b  = blockIdx.y;
  int c0 = blockIdx.x << 4;                     // 48 column tiles
  int m    = lane & 15;
  int half = lane >> 4;
  const unsigned short* arow  = attnbf + ((size_t)(b * 16 + m)) * SEQ;
  const unsigned short* hbase = hb + (size_t)b * SEQ * HIDDEN + c0;
  v8f acc = {0.f, 0.f, 0.f, 0.f, 0.f, 0.f, 0.f, 0.f};

  // stage chunk 0
  {
    const unsigned short* g = hbase + (size_t)lane * HIDDEN;  // s = lane
    v8us w0 = *(const v8us*)(g);
    v8us w1 = *(const v8us*)(g + 8);
#pragma unroll
    for (int j = 0; j < 8; ++j) { tileT[0][j][lane] = w0[j]; tileT[0][8 + j][lane] = w1[j]; }
  }

  for (int kc = 0; kc < 128; ++kc) {
    __syncthreads();                            // single-wave WG: orders LDS (dscnt)
    int buf = kc & 1;
    if (kc + 4 < 128)                           // stream ahead: global_prefetch_b8
      __builtin_prefetch(hbase + (size_t)((kc + 4) * 32 + lane) * HIDDEN, 0, 1);
    if (kc + 1 < 128) {                         // stage next chunk into other buffer
      const unsigned short* g = hbase + (size_t)((kc + 1) * 32 + lane) * HIDDEN;
      v8us w0 = *(const v8us*)(g);
      v8us w1 = *(const v8us*)(g + 8);
      int nb = buf ^ 1;
#pragma unroll
      for (int j = 0; j < 8; ++j) { tileT[nb][j][lane] = w0[j]; tileT[nb][8 + j][lane] = w1[j]; }
    }
    int sA = kc * 32 + half * 8;                // A: attn row m, K table order (8h, 16+8h)
    union { struct { v8us lo, hi; } p; v16bf v; } av;
    av.p.lo = *(const v8us*)(arow + sA);
    av.p.hi = *(const v8us*)(arow + sA + 16);
    union { struct { v8us lo, hi; } p; v16bf v; } bm; // B: col m, K table order (8h, 16+8h) in LDS
    bm.p.lo = *(const v8us*)&tileT[buf][m][half * 8];
    bm.p.hi = *(const v8us*)&tileT[buf][m][half * 8 + 16];
    acc = __builtin_amdgcn_wmma_f32_16x16x32_bf16(false, av.v, false, bm.v, (short)0, acc, false, false);
  }
  int rb = half * 8;
#pragma unroll
  for (int r = 0; r < 8; ++r)
    u[((size_t)(b * 16 + rb + r)) * HIDDEN + c0 + m] = acc[r];
}

// ---- 8) ctx[b][i] = u[b][i/256][:] . Wv[:,i] + bv[i] ----
__global__ void __launch_bounds__(256) k_ctx(const float* __restrict__ u, const float* __restrict__ Wv,
                                             const float* __restrict__ bv, float* __restrict__ ctx) {
  int b = blockIdx.x;
  int i = blockIdx.y * 256 + threadIdx.x;       // 0..3071
  int h = i >> 8;
  const float* ur = u + ((size_t)(b * 16 + h)) * HIDDEN;
  float acc = bv[i];
  for (int c = 0; c < HIDDEN; ++c) acc += ur[c] * Wv[(size_t)c * INTER + i];
  ctx[(size_t)b * INTER + i] = acc;
}

// ---- 9) out = ctx @ Wp + bp; LayerNorm; write pooled ----
__global__ void __launch_bounds__(256) k_out_ln(const float* __restrict__ ctx, const float* __restrict__ Wp,
                                                const float* __restrict__ bp, const float* __restrict__ lnw,
                                                const float* __restrict__ lnb, float* __restrict__ pooled) {
  int b = blockIdx.x, t = threadIdx.x;
  const float* cr = ctx + (size_t)b * INTER;
  float o[3];
#pragma unroll
  for (int r = 0; r < 3; ++r) {
    int j = t + r * 256;
    float acc = bp[j];
    for (int i = 0; i < INTER; ++i) acc += cr[i] * Wp[(size_t)i * HIDDEN + j];
    o[r] = acc;
  }
  __shared__ float red[256];
  red[t] = o[0] + o[1] + o[2]; __syncthreads();
  for (int off = 128; off > 0; off >>= 1) { if (t < off) red[t] += red[t + off]; __syncthreads(); }
  float mean = red[0] * (1.0f / HIDDEN); __syncthreads();
  float d0 = o[0] - mean, d1 = o[1] - mean, d2 = o[2] - mean;
  red[t] = d0 * d0 + d1 * d1 + d2 * d2; __syncthreads();
  for (int off = 128; off > 0; off >>= 1) { if (t < off) red[t] += red[t + off]; __syncthreads(); }
  float inv = rsqrtf(red[0] * (1.0f / HIDDEN) + 1e-6f);
#pragma unroll
  for (int r = 0; r < 3; ++r) {
    int j = t + r * 256;
    pooled[(size_t)b * HIDDEN + j] = (o[r] - mean) * inv * (lnw[j] + 1.0f) + lnb[j];
  }
}

extern "C" void kernel_launch(void* const* d_in, const int* in_sizes, int n_in,
                              void* d_out, int out_size, void* d_ws, size_t ws_size,
                              hipStream_t stream) {
  (void)in_sizes; (void)n_in; (void)out_size; (void)ws_size;
  const float* hs  = (const float*)d_in[0];
  const float* pq  = (const float*)d_in[1];
  const float* Wq  = (const float*)d_in[2];
  const float* bq  = (const float*)d_in[3];
  const float* Wk  = (const float*)d_in[4];
  const float* bk  = (const float*)d_in[5];
  const float* Wv  = (const float*)d_in[6];
  const float* bv  = (const float*)d_in[7];
  const float* Wp  = (const float*)d_in[8];
  const float* bp  = (const float*)d_in[9];
  const float* pds = (const float*)d_in[10];
  const float* lnw = (const float*)d_in[11];
  const float* lnb = (const float*)d_in[12];

  float* out = (float*)d_out;
  char*  ws  = (char*)d_ws;
  float*          qs     = (float*)(ws + WS_QS);
  float*          wke    = (float*)(ws + WS_WKE);
  unsigned short* wkp    = (unsigned short*)(ws + WS_PACK);
  float*          ck     = (float*)(ws + WS_CK);
  float*          logits = (float*)(ws + WS_LOGITS);
  unsigned short* attnbf = (unsigned short*)(ws + WS_ATTNBF);
  float*          u      = (float*)(ws + WS_U);
  float*          ctxp   = (float*)(ws + WS_CTX);
  unsigned short* hb     = (unsigned short*)(ws + WS_HB);

  int n2 = BATCH * SEQ * HIDDEN / 2;
  k_cvt_hidden<<<(n2 + 255) / 256, 256, 0, stream>>>(hs, (unsigned int*)hb, n2);
  k_qproj<<<INTER / 256, 256, 0, stream>>>(pq, Wq, bq, pds, qs);
  k_wkeff<<<(HIDDEN * 16) / 256, 256, 0, stream>>>(Wk, qs, wke);
  k_pack<<<(24 * 32 * 16) / 256, 256, 0, stream>>>(wke, bk, qs, wkp, ck);
  k_logits<<<(BATCH * SEQ / 16) / 8, 256, 0, stream>>>(hb, wkp, ck, logits);
  k_softmax<<<BATCH * 16, 256, 0, stream>>>(logits, out + BATCH * HIDDEN, attnbf);
  k_ugemm<<<dim3(48, BATCH), 32, 0, stream>>>(hb, attnbf, u);
  k_ctx<<<dim3(BATCH, INTER / 256), 256, 0, stream>>>(u, Wv, bv, ctxp);
  k_out_ln<<<BATCH, 256, 0, stream>>>(ctxp, Wp, bp, lnw, lnb, out);
}
